// KNRM_49065706390009
// MI455X (gfx1250) — matrix operand order repaired and checked by reading, not compiled, not run on hardware
//
#include <hip/hip_runtime.h>
#include <hip/hip_fp16.h>

typedef _Float16 v16h __attribute__((ext_vector_type(16)));
typedef _Float16 v8h  __attribute__((ext_vector_type(8)));
typedef float    v8f  __attribute__((ext_vector_type(8)));

#define VOCAB 100000
#define EDIM  128
#define NB    1024
#define QLEN  64
#define DLEN  512
#define NK    21

// ---------------------------------------------------------------------------
// Kernel 1: f32 -> f16 embedding table + per-vocab reciprocal norms.
// rnorm[0] == 0 (padding row is all-zero) reproduces the max(..,eps) clamp:
// any product involving a zero-norm row yields M == 0, matching the reference.
// ---------------------------------------------------------------------------
__global__ __launch_bounds__(128) void knrm_prep(
    const float* __restrict__ emb, _Float16* __restrict__ eh,
    float* __restrict__ rn)
{
    const int v = blockIdx.x;
    const int t = threadIdx.x;
    __shared__ float red[128];
    const float x = emb[(size_t)v * EDIM + t];
    eh[(size_t)v * EDIM + t] = (_Float16)x;
    red[t] = x * x;
    __syncthreads();
    #pragma unroll
    for (int s = 64; s > 0; s >>= 1) {
        if (t < s) red[t] += red[t + s];
        __syncthreads();
    }
    if (t == 0) rn[v] = (red[0] > 0.0f) ? rsqrtf(red[0]) : 0.0f;
}

// ---------------------------------------------------------------------------
// Kernel 2: fused cosine-sim WMMA GEMM + Gaussian kernel pooling.
// One workgroup = one (batch, pair). 8 waves; wave w owns query tile (w&3)
// and doc tiles [ (w>>2)*16, (w>>2)*16+16 ).
// C tile is M^T: rows = doc (reduced), cols = query (kept per lane).
// ---------------------------------------------------------------------------
__global__ __launch_bounds__(256) void knrm_scores(
    const int* __restrict__ q1, const int* __restrict__ d1,
    const int* __restrict__ q2, const int* __restrict__ d2,
    const _Float16* __restrict__ eh, const float* __restrict__ rnorm,
    float* __restrict__ feats /* [2][NB][NK] */)
{
    const int blk  = blockIdx.x;
    const int pair = blk >> 10;
    const int b    = blk & 1023;
    const int* __restrict__ qids_g = pair ? q2 : q1;
    const int* __restrict__ dids_g = pair ? d2 : d1;

    __shared__ float sums[QLEN * NK];
    for (int i = threadIdx.x; i < QLEN * NK; i += 256) sums[i] = 0.0f;
    __syncthreads();

    const int wave = threadIdx.x >> 5;
    const int lane = threadIdx.x & 31;
    const int qt   = wave & 3;   // query tile 0..3
    const int half = wave >> 2;  // which 16 doc tiles
    const int n    = lane & 15;  // query column within tile
    const int hi   = lane >> 4;  // lane half (selects K range / C row half)

    // --- B operand (query embeddings), loaded once, lives in VGPRs -------
    // B in contiguous K order (this code's choice): lane holds 16 contiguous K (lanes 0-15: K 0-15,
    // lanes 16-31: K 16-31) of column n.
    const int qid = qids_g[b * QLEN + qt * 16 + n];
    const _Float16* qrow = eh + (size_t)qid * EDIM;
    const float rq = rnorm[qid];

    v16h Bop[4];
    #pragma unroll
    for (int w = 0; w < 4; ++w)
        Bop[w] = *(const v16h*)(qrow + w * 32 + hi * 16);

    float acc[NK];
    #pragma unroll
    for (int k = 0; k < NK; ++k) acc[k] = 0.0f;

    const int am = lane & 15;              // A-operand row for this lane
    const float KEXP = -72.13475204f;      // -50 * log2(e)  (sigma = 0.1)

    for (int t = 0; t < 16; ++t) {
        const int dt    = half * 16 + t;
        const int dbase = b * DLEN + dt * 16;
        if (t < 15) __builtin_prefetch(dids_g + dbase + 16, 0, 3);

        // --- A operand (doc embeddings) ------------------------------
        // A in contiguous K order (this code's choice): lane = row m, lanes 0-15 hold K 0..15,
        // lanes 16-31 hold K 16..31 (+w*32), contiguous like B.
        const int aid = dids_g[dbase + am];
        const _Float16* arow = eh + (size_t)aid * EDIM;

        v8f c = {0.f, 0.f, 0.f, 0.f, 0.f, 0.f, 0.f, 0.f};
        #pragma unroll
        for (int w = 0; w < 4; ++w) {
            v8h lo = *(const v8h*)(arow + w * 32 + hi * 16);
            v8h hp = *(const v8h*)(arow + w * 32 + hi * 16 + 8);
            v16h a;
            #pragma unroll
            for (int i = 0; i < 8; ++i) { a[i] = lo[i]; a[i + 8] = hp[i]; }
            c = __builtin_amdgcn_wmma_f32_16x16x32_f16(
                    false, a, false, Bop[w], (short)0, c, false, false);
        }

        // --- fused Gaussian kernel pooling on the accumulator tile ----
        // lane's 8 C values: rows m = i + hi*8 (doc), column n (query).
        const int* dp = dids_g + dbase + hi * 8;
        int rid[8];
        {
            const int4 r0 = *(const int4*)(dp);
            const int4 r1 = *(const int4*)(dp + 4);
            rid[0] = r0.x; rid[1] = r0.y; rid[2] = r0.z; rid[3] = r0.w;
            rid[4] = r1.x; rid[5] = r1.y; rid[6] = r1.z; rid[7] = r1.w;
        }

        #pragma unroll
        for (int i = 0; i < 8; ++i) {
            const float rd = rnorm[rid[i]];
            const float m  = c[i] * rq * rd;     // cosine similarity
            #pragma unroll
            for (int k = 0; k < 20; ++k) {
                const float mu = -0.95f + 0.1f * (float)k;
                const float dd = m - mu;
                acc[k] += __builtin_amdgcn_exp2f(KEXP * dd * dd);
            }
            // exact-match kernel (mu=1, sigma=1e-3): fires iff ids match
            // and the token is not padding (cos==1 exactly in f32 ref).
            acc[20] += ((rid[i] == qid) & (qid != 0)) ? 1.0f : 0.0f;
        }
    }

    // --- reduce over doc dimension across lanes/waves via LDS atomics ----
    float* srow = &sums[(qt * 16 + n) * NK];
    #pragma unroll
    for (int k = 0; k < NK; ++k) atomicAdd(&srow[k], acc[k]);
    __syncthreads();

    // --- log1p over doc-sums, reduce over query dimension ----------------
    if (threadIdx.x < NK) {
        const int k = threadIdx.x;
        float s = 0.0f;
        for (int q = 0; q < QLEN; ++q)
            s += 0.69314718056f * __builtin_amdgcn_logf(1.0f + sums[q * NK + k]);
        feats[((size_t)pair * NB + b) * NK + k] = s;
    }
}

// ---------------------------------------------------------------------------
// Kernel 3: per-batch 21->10->5->1 MLP on both pairs + sigmoid(l1-l2).
// ---------------------------------------------------------------------------
__global__ __launch_bounds__(256) void knrm_mlp(
    const float* __restrict__ feats,
    const float* __restrict__ W0, const float* __restrict__ b0,
    const float* __restrict__ W1, const float* __restrict__ b1,
    const float* __restrict__ W2, const float* __restrict__ b2,
    float* __restrict__ out)
{
    const int b = blockIdx.x * blockDim.x + threadIdx.x;
    if (b >= NB) return;

    float logit[2];
    #pragma unroll
    for (int p = 0; p < 2; ++p) {
        const float* f = feats + ((size_t)p * NB + b) * NK;
        float h0[10];
        #pragma unroll
        for (int j = 0; j < 10; ++j) {
            float s = b0[j];
            #pragma unroll
            for (int k = 0; k < NK; ++k) s += f[k] * W0[j * NK + k];
            h0[j] = s > 0.0f ? s : 0.0f;
        }
        float h1[5];
        #pragma unroll
        for (int j = 0; j < 5; ++j) {
            float s = b1[j];
            #pragma unroll
            for (int k = 0; k < 10; ++k) s += h0[k] * W1[j * 10 + k];
            h1[j] = s > 0.0f ? s : 0.0f;
        }
        float l = b2[0];
        #pragma unroll
        for (int k = 0; k < 5; ++k) l += h1[k] * W2[k];
        logit[p] = l;
    }
    const float z = logit[0] - logit[1];
    out[b] = 1.0f / (1.0f + __builtin_amdgcn_exp2f(-1.44269504f * z));
}

// ---------------------------------------------------------------------------
extern "C" void kernel_launch(void* const* d_in, const int* in_sizes, int n_in,
                              void* d_out, int out_size, void* d_ws, size_t ws_size,
                              hipStream_t stream) {
    (void)in_sizes; (void)n_in; (void)out_size; (void)ws_size;

    const int*   q1  = (const int*)  d_in[0];
    const int*   d1  = (const int*)  d_in[1];
    const int*   q2  = (const int*)  d_in[2];
    const int*   d2  = (const int*)  d_in[3];
    const float* emb = (const float*)d_in[4];
    const float* W0  = (const float*)d_in[5];
    const float* b0  = (const float*)d_in[6];
    const float* W1  = (const float*)d_in[7];
    const float* b1  = (const float*)d_in[8];
    const float* W2  = (const float*)d_in[9];
    const float* b2  = (const float*)d_in[10];

    // workspace layout
    char* ws = (char*)d_ws;
    _Float16* eh    = (_Float16*)(ws);              // 100000*128*2 = 25,600,000 B
    float*    rn    = (float*)(ws + 25600000);      // 100000*4     =    400,000 B
    float*    feats = (float*)(ws + 26000000);      // 2*1024*21*4  =    172,032 B

    knrm_prep  <<<VOCAB, 128, 0, stream>>>(emb, eh, rn);
    knrm_scores<<<2 * NB, 256, 0, stream>>>(q1, d1, q2, d2, eh, rn, feats);
    knrm_mlp   <<<(NB + 255) / 256, 256, 0, stream>>>(feats, W0, b0, W1, b1, W2, b2,
                                                      (float*)d_out);
}
